// CoLightNet_83133386981784
// MI455X (gfx1250) — matrix operand order repaired and checked by reading, not compiled, not run on hardware
//
#include <hip/hip_runtime.h>
#include <hip/hip_bf16.h>

// ---- problem constants (match reference) ----
#define NNODES  5000
#define RTOT    160000            // B*N rows, divisible by 16
#define KNBR    8
#define HEADS   2

// ---- WMMA fragment types (CDNA5 wave32) ----
typedef __attribute__((ext_vector_type(16))) __bf16         v16bf;
typedef __attribute__((ext_vector_type(16))) unsigned short v16u;
typedef __attribute__((ext_vector_type(8)))  float          v8f;

// Native f32 -> bf16 (RTNE) conversion; lets the compiler emit v_cvt ops
// instead of a manual bfe/add3/shift chain.
__device__ __forceinline__ unsigned short f2bf(float f) {
    __bf16 h = (__bf16)f;
    return __builtin_bit_cast(unsigned short, h);
}
__device__ __forceinline__ void unpack2(unsigned w, float& a, float& b) {
    a = __builtin_bit_cast(float, w << 16);
    b = __builtin_bit_cast(float, w & 0xFFFF0000u);
}

union FragCaster { v16u u; uint4 q[2]; };

// A fragment (16x32 bf16 MxK): lane = row; lanes 0-15 take K +0..7 & +16..23
// of base, lanes 16-31 (base already offset by half*8) take +8..15 & +24..31.
__device__ __forceinline__ v16u load_a_bf16(const unsigned short* base) {
    FragCaster fc;
    fc.q[0] = *(const uint4*)(base);
    fc.q[1] = *(const uint4*)(base + 16);
    return fc.u;
}
__device__ __forceinline__ v16u load_a_f32(const float* base) {
    const float4* p = (const float4*)base;
    float4 x0 = p[0], x1 = p[1], x2 = p[4], x3 = p[5];
    v16u u;
    u[0]=f2bf(x0.x); u[1]=f2bf(x0.y); u[2]=f2bf(x0.z); u[3]=f2bf(x0.w);
    u[4]=f2bf(x1.x); u[5]=f2bf(x1.y); u[6]=f2bf(x1.z); u[7]=f2bf(x1.w);
    u[8]=f2bf(x2.x); u[9]=f2bf(x2.y); u[10]=f2bf(x2.z); u[11]=f2bf(x2.w);
    u[12]=f2bf(x3.x); u[13]=f2bf(x3.y); u[14]=f2bf(x3.z); u[15]=f2bf(x3.w);
    return u;
}
// B fragment from LDS-transposed weights: K +0..7 & +16..23 of base (base offset by half*8).
__device__ __forceinline__ v16u load_b16(const unsigned short* p) {
    FragCaster fc;
    fc.q[0] = *(const uint4*)(p);
    fc.q[1] = *(const uint4*)(p + 16);
    return fc.u;
}
__device__ __forceinline__ v8f wmma_bf16(v16u a, v16u b, v8f c) {
    return __builtin_amdgcn_wmma_f32_16x16x32_bf16(
        false, __builtin_bit_cast(v16bf, a),
        false, __builtin_bit_cast(v16bf, b),
        (short)0, c, false, false);
}
__device__ __forceinline__ v8f bias_acc(float b) {
    v8f a = {b, b, b, b, b, b, b, b};
    return a;
}
// D tile -> per-wave LDS bounce (row-major bf16, stride 40 keeps 16B align).
__device__ __forceinline__ void store_tile(unsigned short (*tile)[40], int l15,
                                           int halfSel, int nt, v8f acc, bool relu) {
#pragma unroll
    for (int j = 0; j < 8; ++j) {
        float v = acc[j];
        if (relu) v = fmaxf(v, 0.0f);
        tile[j + halfSel * 8][nt * 16 + l15] = f2bf(v);
    }
}

// ---------------------------------------------------------------------------
// Kernel 1: features -> relu(X@We1) -> relu(@We2) -> fused Q/K/V projections.
// h1/h2 live only in per-wave LDS tiles. Q f32, K/V bf16 (gathered 8x later).
// ---------------------------------------------------------------------------
__global__ __launch_bounds__(256)
void embed_qkv_kernel(const float* __restrict__ X,
                      const float* __restrict__ We1, const float* __restrict__ be1,
                      const float* __restrict__ We2, const float* __restrict__ be2,
                      const float* __restrict__ Wq,  const float* __restrict__ bq,
                      const float* __restrict__ Wk,  const float* __restrict__ bk,
                      const float* __restrict__ Wv,  const float* __restrict__ bv,
                      float* __restrict__ Qf, unsigned short* __restrict__ Kb,
                      unsigned short* __restrict__ Vb)
{
    __shared__ __align__(16) unsigned short W1t[32][72];      // (64x32) transposed
    __shared__ __align__(16) unsigned short W2t[32][40];      // (32x32) transposed
    __shared__ __align__(16) unsigned short W3t[3][32][40];   // Wq/Wk/Wv transposed
    __shared__ float b1l[32], b2l[32], b3l[3][32];
    __shared__ __align__(16) unsigned short tileA[8][16][40];
    __shared__ __align__(16) unsigned short tileB[8][16][40];

    const int tid = threadIdx.x;
    for (int idx = tid; idx < 32 * 64; idx += 256) {
        const int n = idx >> 6, k = idx & 63;
        W1t[n][k] = f2bf(We1[k * 32 + n]);
    }
    for (int idx = tid; idx < 32 * 32; idx += 256) {
        const int n = idx >> 5, k = idx & 31;
        W2t[n][k] = f2bf(We2[k * 32 + n]);
    }
    const float* Ws[3] = {Wq, Wk, Wv};
    const float* bs[3] = {bq, bk, bv};
    for (int idx = tid; idx < 3 * 32 * 32; idx += 256) {
        const int m = idx >> 10, rem = idx & 1023;
        const int n = rem >> 5, k = rem & 31;
        W3t[m][n][k] = f2bf(Ws[m][k * 32 + n]);
    }
    if (tid < 32) { b1l[tid] = be1[tid]; b2l[tid] = be2[tid]; }
    if (tid < 96) b3l[tid >> 5][tid & 31] = bs[tid >> 5][tid & 31];
    __syncthreads();

    const int lane = tid & 31, l15 = lane & 15, halfSel = lane >> 4;
    const int w = tid >> 5;
    const int row0 = (blockIdx.x * 8 + w) * 16;

    // GEMM1 (DIN=64): X tile -> relu -> tileA
    const float* Xr = X + (size_t)(row0 + l15) * 64 + halfSel * 8;
    const v16u a0 = load_a_f32(Xr);
    const v16u a1 = load_a_f32(Xr + 32);
#pragma unroll
    for (int nt = 0; nt < 2; ++nt) {
        const int col = nt * 16 + l15;
        v8f acc = bias_acc(b1l[col]);
        acc = wmma_bf16(a0, load_b16(&W1t[col][halfSel * 8]), acc);
        acc = wmma_bf16(a1, load_b16(&W1t[col][32 + halfSel * 8]), acc);
        store_tile(tileA[w], l15, halfSel, nt, acc, true);
    }
    __syncthreads();

    // GEMM2 (DIN=32): tileA -> relu -> tileB
    const v16u a2 = load_a_bf16(&tileA[w][l15][halfSel * 8]);
#pragma unroll
    for (int nt = 0; nt < 2; ++nt) {
        const int col = nt * 16 + l15;
        v8f acc = bias_acc(b2l[col]);
        acc = wmma_bf16(a2, load_b16(&W2t[col][halfSel * 8]), acc);
        store_tile(tileB[w], l15, halfSel, nt, acc, true);
    }
    __syncthreads();

    // Fused Q/K/V from tileB
    const v16u a3 = load_a_bf16(&tileB[w][l15][halfSel * 8]);
#pragma unroll
    for (int m = 0; m < 3; ++m) {
#pragma unroll
        for (int nt = 0; nt < 2; ++nt) {
            const int col = nt * 16 + l15;
            v8f acc = bias_acc(b3l[m][col]);
            acc = wmma_bf16(a3, load_b16(&W3t[m][col][halfSel * 8]), acc);
#pragma unroll
            for (int j = 0; j < 8; ++j) {
                const size_t o = (size_t)(row0 + j + halfSel * 8) * 32 + col;
                if (m == 0)      Qf[o] = acc[j];
                else if (m == 1) Kb[o] = f2bf(acc[j]);
                else             Vb[o] = f2bf(acc[j]);
            }
        }
    }
}

// ---------------------------------------------------------------------------
// Kernel 3: o -> relu(o@Wo+bo) -> fused Q/K/V of the next attention layer.
// ---------------------------------------------------------------------------
__global__ __launch_bounds__(256)
void proj_qkv_kernel(const unsigned short* __restrict__ O,
                     const float* __restrict__ Wo, const float* __restrict__ bo,
                     const float* __restrict__ Wq, const float* __restrict__ bq,
                     const float* __restrict__ Wk, const float* __restrict__ bk,
                     const float* __restrict__ Wv, const float* __restrict__ bv,
                     float* __restrict__ Qf, unsigned short* __restrict__ Kb,
                     unsigned short* __restrict__ Vb)
{
    __shared__ __align__(16) unsigned short Wot[32][40];
    __shared__ __align__(16) unsigned short W3t[3][32][40];
    __shared__ float bol[32], b3l[3][32];
    __shared__ __align__(16) unsigned short tile[8][16][40];

    const int tid = threadIdx.x;
    for (int idx = tid; idx < 32 * 32; idx += 256) {
        const int n = idx >> 5, k = idx & 31;
        Wot[n][k] = f2bf(Wo[k * 32 + n]);
    }
    const float* Ws[3] = {Wq, Wk, Wv};
    const float* bs[3] = {bq, bk, bv};
    for (int idx = tid; idx < 3 * 32 * 32; idx += 256) {
        const int m = idx >> 10, rem = idx & 1023;
        const int n = rem >> 5, k = rem & 31;
        W3t[m][n][k] = f2bf(Ws[m][k * 32 + n]);
    }
    if (tid < 32) bol[tid] = bo[tid];
    if (tid < 96) b3l[tid >> 5][tid & 31] = bs[tid >> 5][tid & 31];
    __syncthreads();

    const int lane = tid & 31, l15 = lane & 15, halfSel = lane >> 4;
    const int w = tid >> 5;
    const int row0 = (blockIdx.x * 8 + w) * 16;

    const v16u a = load_a_bf16(O + (size_t)(row0 + l15) * 32 + halfSel * 8);
#pragma unroll
    for (int nt = 0; nt < 2; ++nt) {
        const int col = nt * 16 + l15;
        v8f acc = bias_acc(bol[col]);
        acc = wmma_bf16(a, load_b16(&Wot[col][halfSel * 8]), acc);
        store_tile(tile[w], l15, halfSel, nt, acc, true);
    }
    __syncthreads();

    const v16u a2 = load_a_bf16(&tile[w][l15][halfSel * 8]);
#pragma unroll
    for (int m = 0; m < 3; ++m) {
#pragma unroll
        for (int nt = 0; nt < 2; ++nt) {
            const int col = nt * 16 + l15;
            v8f acc = bias_acc(b3l[m][col]);
            acc = wmma_bf16(a2, load_b16(&W3t[m][col][halfSel * 8]), acc);
#pragma unroll
            for (int j = 0; j < 8; ++j) {
                const size_t o = (size_t)(row0 + j + halfSel * 8) * 32 + col;
                if (m == 0)      Qf[o] = acc[j];
                else if (m == 1) Kb[o] = f2bf(acc[j]);
                else             Vb[o] = f2bf(acc[j]);
            }
        }
    }
}

// ---------------------------------------------------------------------------
// Kernel 5: o1 -> relu(o1@Wo1+bo1) -> q = h4@Wqh+bqh (dout=8, padded tile).
// ---------------------------------------------------------------------------
__global__ __launch_bounds__(256)
void out_head_kernel(const unsigned short* __restrict__ O,
                     const float* __restrict__ Wo, const float* __restrict__ bo,
                     const float* __restrict__ Wh, const float* __restrict__ bh,
                     float* __restrict__ out_q)
{
    __shared__ __align__(16) unsigned short Wot[32][40];
    __shared__ __align__(16) unsigned short Wht[16][40];    // (32x8) padded to 16 cols
    __shared__ float bol[32], bhl[16];
    __shared__ __align__(16) unsigned short tile[8][16][40];

    const int tid = threadIdx.x;
    for (int idx = tid; idx < 32 * 32; idx += 256) {
        const int n = idx >> 5, k = idx & 31;
        Wot[n][k] = f2bf(Wo[k * 32 + n]);
    }
    for (int idx = tid; idx < 16 * 32; idx += 256) {
        const int n = idx >> 5, k = idx & 31;
        Wht[n][k] = (n < 8) ? f2bf(Wh[k * 8 + n]) : (unsigned short)0;
    }
    if (tid < 32) bol[tid] = bo[tid];
    if (tid < 16) bhl[tid] = (tid < 8) ? bh[tid] : 0.0f;
    __syncthreads();

    const int lane = tid & 31, l15 = lane & 15, halfSel = lane >> 4;
    const int w = tid >> 5;
    const int row0 = (blockIdx.x * 8 + w) * 16;

    const v16u a = load_a_bf16(O + (size_t)(row0 + l15) * 32 + halfSel * 8);
#pragma unroll
    for (int nt = 0; nt < 2; ++nt) {
        const int col = nt * 16 + l15;
        v8f acc = bias_acc(bol[col]);
        acc = wmma_bf16(a, load_b16(&Wot[col][halfSel * 8]), acc);
        store_tile(tile[w], l15, halfSel, nt, acc, true);
    }
    __syncthreads();

    const v16u a2 = load_a_bf16(&tile[w][l15][halfSel * 8]);
    {
        const int col = l15;                       // single padded 16-col tile
        v8f acc = bias_acc(bhl[col]);
        acc = wmma_bf16(a2, load_b16(&Wht[col][halfSel * 8]), acc);
        if (col < 8) {
#pragma unroll
            for (int j = 0; j < 8; ++j)
                out_q[(size_t)(row0 + j + halfSel * 8) * 8 + col] = acc[j];
        }
    }
}

// ---------------------------------------------------------------------------
// Neighbor attention: thread = node, both heads. Q f32, K/V bf16 (gathered).
// ---------------------------------------------------------------------------
__global__ __launch_bounds__(256)
void att_kernel(const float* __restrict__ Q, const unsigned short* __restrict__ Kb,
                const unsigned short* __restrict__ Vb, const int* __restrict__ adj,
                float* __restrict__ att_out, unsigned short* __restrict__ o_out)
{
    const int r = blockIdx.x * blockDim.x + threadIdx.x;
    if (r >= RTOT) return;
    const int rowBase = (r / NNODES) * NNODES;

    float qv[32];
    const float4* qp = (const float4*)(Q + (size_t)r * 32);
#pragma unroll
    for (int i = 0; i < 8; ++i) {
        float4 t = qp[i];
        qv[4*i+0] = t.x; qv[4*i+1] = t.y; qv[4*i+2] = t.z; qv[4*i+3] = t.w;
    }
    const int4* ap = (const int4*)(adj + (size_t)r * KNBR);
    const int4 aa = ap[0], ab = ap[1];
    const int nb[KNBR] = {rowBase + aa.x, rowBase + aa.y, rowBase + aa.z, rowBase + aa.w,
                          rowBase + ab.x, rowBase + ab.y, rowBase + ab.z, rowBase + ab.w};

    float s[HEADS][KNBR];
#pragma unroll
    for (int k = 0; k < KNBR; ++k) {
        const uint4* kp = (const uint4*)(Kb + (size_t)nb[k] * 32);
        float s0 = 0.0f, s1 = 0.0f;
#pragma unroll
        for (int i = 0; i < 4; ++i) {
            uint4 t = kp[i];
            float f0,f1,f2,f3,f4,f5,f6,f7;
            unpack2(t.x, f0, f1); unpack2(t.y, f2, f3);
            unpack2(t.z, f4, f5); unpack2(t.w, f6, f7);
            const float d = qv[8*i+0]*f0 + qv[8*i+1]*f1 + qv[8*i+2]*f2 + qv[8*i+3]*f3
                          + qv[8*i+4]*f4 + qv[8*i+5]*f5 + qv[8*i+6]*f6 + qv[8*i+7]*f7;
            if (i < 2) s0 += d; else s1 += d;
        }
        s[0][k] = s0 * 0.25f;                      // 1/sqrt(DV)
        s[1][k] = s1 * 0.25f;
    }

    float p[HEADS][KNBR];
#pragma unroll
    for (int h = 0; h < HEADS; ++h) {
        float mx = s[h][0];
#pragma unroll
        for (int k = 1; k < KNBR; ++k) mx = fmaxf(mx, s[h][k]);
        float sum = 0.0f;
#pragma unroll
        for (int k = 0; k < KNBR; ++k) { float e = __expf(s[h][k] - mx); p[h][k] = e; sum += e; }
        const float inv = 1.0f / sum;
#pragma unroll
        for (int k = 0; k < KNBR; ++k) p[h][k] *= inv;
    }
    float4* atp = (float4*)(att_out + (size_t)r * HEADS * KNBR);
    atp[0] = make_float4(p[0][0], p[0][1], p[0][2], p[0][3]);
    atp[1] = make_float4(p[0][4], p[0][5], p[0][6], p[0][7]);
    atp[2] = make_float4(p[1][0], p[1][1], p[1][2], p[1][3]);
    atp[3] = make_float4(p[1][4], p[1][5], p[1][6], p[1][7]);

    float acc[32];
#pragma unroll
    for (int c = 0; c < 32; ++c) acc[c] = 0.0f;
#pragma unroll
    for (int k = 0; k < KNBR; ++k) {
        const uint4* vp = (const uint4*)(Vb + (size_t)nb[k] * 32);
#pragma unroll
        for (int i = 0; i < 4; ++i) {
            uint4 t = vp[i];
            const float wgt = p[i >> 1][k];
            float f0,f1,f2,f3,f4,f5,f6,f7;
            unpack2(t.x, f0, f1); unpack2(t.y, f2, f3);
            unpack2(t.z, f4, f5); unpack2(t.w, f6, f7);
            acc[8*i+0] += wgt*f0; acc[8*i+1] += wgt*f1; acc[8*i+2] += wgt*f2; acc[8*i+3] += wgt*f3;
            acc[8*i+4] += wgt*f4; acc[8*i+5] += wgt*f5; acc[8*i+6] += wgt*f6; acc[8*i+7] += wgt*f7;
        }
    }
    unsigned ow[8];
#pragma unroll
    for (int j = 0; j < 8; ++j)
        ow[j] = (unsigned)f2bf(acc[2*j]) | ((unsigned)f2bf(acc[2*j+1]) << 16);
    uint4* op = (uint4*)(o_out + (size_t)r * 32);
    op[0] = make_uint4(ow[0], ow[1], ow[2], ow[3]);
    op[1] = make_uint4(ow[4], ow[5], ow[6], ow[7]);
}

// ---------------------------------------------------------------------------
extern "C" void kernel_launch(void* const* d_in, const int* in_sizes, int n_in,
                              void* d_out, int out_size, void* d_ws, size_t ws_size,
                              hipStream_t stream) {
    (void)in_sizes; (void)n_in; (void)out_size; (void)ws_size;
    const float* features = (const float*)d_in[0];
    const int*   adj      = (const int*)d_in[1];
    const float* We1 = (const float*)d_in[2],  *be1 = (const float*)d_in[3];
    const float* We2 = (const float*)d_in[4],  *be2 = (const float*)d_in[5];
    const float* Wq0 = (const float*)d_in[6],  *bq0 = (const float*)d_in[7];
    const float* Wk0 = (const float*)d_in[8],  *bk0 = (const float*)d_in[9];
    const float* Wv0 = (const float*)d_in[10], *bv0 = (const float*)d_in[11];
    const float* Wo0 = (const float*)d_in[12], *bo0 = (const float*)d_in[13];
    const float* Wq1 = (const float*)d_in[14], *bq1 = (const float*)d_in[15];
    const float* Wk1 = (const float*)d_in[16], *bk1 = (const float*)d_in[17];
    const float* Wv1 = (const float*)d_in[18], *bv1 = (const float*)d_in[19];
    const float* Wo1 = (const float*)d_in[20], *bo1 = (const float*)d_in[21];
    const float* Wqh = (const float*)d_in[22], *bqh = (const float*)d_in[23];

    // workspace: Qf f32 20.48MB + Kb/Vb/bufO bf16 10.24MB each = 51.2MB
    char* ws = (char*)d_ws;
    float*          Qf   = (float*)         (ws);
    unsigned short* Kb   = (unsigned short*)(ws + 20480000);
    unsigned short* Vb   = (unsigned short*)(ws + 30720000);
    unsigned short* bufO = (unsigned short*)(ws + 40960000);

    float* out_q    = (float*)d_out;            // (B,N,8)
    float* out_att0 = out_q + 1280000;          // (B,N,2,8)
    float* out_att1 = out_q + 3840000;          // (B,N,2,8)

    const int GB = RTOT / (16 * 8);             // 1250 blocks, wave = 16-row tile
    const int AB = RTOT / 256;                  // 625 blocks, thread = node

    embed_qkv_kernel<<<GB,256,0,stream>>>(features, We1, be1, We2, be2,
                                          Wq0, bq0, Wk0, bk0, Wv0, bv0, Qf, Kb, Vb);
    att_kernel<<<AB,256,0,stream>>>(Qf, Kb, Vb, adj, out_att0, bufO);
    proj_qkv_kernel<<<GB,256,0,stream>>>(bufO, Wo0, bo0,
                                         Wq1, bq1, Wk1, bk1, Wv1, bv1, Qf, Kb, Vb);
    att_kernel<<<AB,256,0,stream>>>(Qf, Kb, Vb, adj, out_att1, bufO);
    out_head_kernel<<<GB,256,0,stream>>>(bufO, Wo1, bo1, Wqh, bqh, out_q);
}
